// KAME_2293512536390
// MI455X (gfx1250) — matrix operand order repaired and checked by reading, not compiled, not run on hardware
//
#include <hip/hip_runtime.h>
#include <hip/hip_bf16.h>
#include <math.h>

// ---- model dims ----
#define BB   128
#define TT   50
#define MM   (BB * TT)     // 6400
#define LEAF 4880
#define AA   64
#define ATTN 128
#define RNN  128
#define G3   (3 * RNN)     // 384
#define NC   283
#define NCP  288           // padded to 18 tiles of 16
#define KCH  64            // proj K-chunk (2 WMMA steps per stage)

typedef __attribute__((ext_vector_type(16))) _Float16 v16h;
typedef __attribute__((ext_vector_type(8)))  _Float16 v8h;
typedef __attribute__((ext_vector_type(4)))  _Float16 v4h;
typedef __attribute__((ext_vector_type(8)))  float    v8f;

#define WMMA_F16(a, b, c) \
    __builtin_amdgcn_wmma_f32_16x16x32_f16(false, (a), false, (b), (short)0, (c), false, false)

__device__ __forceinline__ float sigmoidf_(float x) {
    return 1.f / (1.f + __expf(-x));
}

__device__ __forceinline__ v16h make_frag(v8h lo, v8h hi) {
    v16h o;
#pragma unroll
    for (int i = 0; i < 8; ++i) { o[i] = lo[i]; o[i + 8] = hi[i]; }
    return o;
}

// ---------------------------------------------------------------------------
// WMMA fragment loaders (wave32, 16x16x32 f16 -> f32), fully vectorized.
// A 16x32 (MxK): lanes 0-15 rows, per-lane halves = K {base..base+7, base+16..base+23},
//                base = k0 + 8*(lane>=16). Two 16B loads.
// Requires: ldA*2B and k0*2B multiples of 16B.
// ---------------------------------------------------------------------------
__device__ __forceinline__ v16h load_a_frag_v(const _Float16* A, int ldA,
                                              int m0, int k0) {
    int l = threadIdx.x & 31;
    const _Float16* row =
        A + (size_t)(m0 + (l & 15)) * ldA + k0 + ((l >> 4) << 3);
    v8h lo = *(const v8h*)(row);
    v8h hi = *(const v8h*)(row + 16);
    return make_frag(lo, hi);
}

// B 32x16 (KxN) from row-major [K,N] source: lane = N col n0+(l&15), halves =
// K {base..base+7, base+16..base+23}, base = k0 + 8*(lane>=16). Strided loads.
__device__ __forceinline__ v16h load_b_frag_v(const _Float16* Bm, int ldB,
                                              int k0, int n0) {
    int l = threadIdx.x & 31;
    const _Float16* p = Bm + (size_t)(k0 + ((l >> 4) << 3)) * ldB + n0 + (l & 15);
    v16h out;
#pragma unroll
    for (int i = 0; i < 8; ++i) {
        out[i] = p[(size_t)i * ldB];
        out[i + 8] = p[(size_t)(16 + i) * ldB];
    }
    return out;
}

// checked variant (proj K tail only)
__device__ __forceinline__ v16h load_b_frag_chk(const _Float16* Bm, int ldB,
                                                int k0, int n0, int kmax) {
    int l = threadIdx.x & 31;
    int kb = k0 + ((l >> 4) << 3);
    const _Float16* p = Bm + n0 + (l & 15);
    v16h out;
#pragma unroll
    for (int i = 0; i < 16; ++i) {
        int k = kb + (i < 8 ? i : i + 8);
        out[i] = (k < kmax) ? p[(size_t)k * ldB] : (_Float16)0.f;
    }
    return out;
}

// ---------------------------------------------------------------------------
// 0a) plain f32 -> f16
// ---------------------------------------------------------------------------
__global__ void cvt_f32_f16_kernel(const float* __restrict__ src,
                                   _Float16* __restrict__ dst, int n) {
    int i = blockIdx.x * blockDim.x + threadIdx.x;
    if (i < n) dst[i] = (_Float16)src[i];
}

// 0b) transpose-convert: src [N,K] f32 -> dst [K,Npad] f16 (zero-padded cols)
__global__ void cvt_transpose_kernel(const float* __restrict__ src,
                                     _Float16* __restrict__ dst,
                                     int N, int K, int Npad) {
    int i = blockIdx.x * blockDim.x + threadIdx.x;
    if (i >= K * Npad) return;
    int k = i / Npad, n = i - k * Npad;
    dst[i] = (n < N) ? (_Float16)src[(size_t)n * K + k] : (_Float16)0.f;
}

// ---------------------------------------------------------------------------
// 1) x = tanh(inputs_x @ dag_emb)   [MM,LEAF]x[LEAF,ATTN] -> f16 [MM,ATTN]
//    block 256 (8 waves): 16-row strip, wave w owns N-tile w.
//    Double-buffered vectorized LDS staging of A (f32->f16).
// ---------------------------------------------------------------------------
__global__ void proj_tanh_kernel(const float* __restrict__ X,
                                 const _Float16* __restrict__ Wkn,
                                 _Float16* __restrict__ out) {
    __shared__ _Float16 aT[2][16 * KCH];
    const int tid  = threadIdx.x;
    const int wave = tid >> 5;
    const int m0   = blockIdx.x * 16;
    const int n0   = wave * 16;

    // staging map: thread -> (row, 4-col group) of the 16x64 chunk
    const int srow = tid >> 4;          // 0..15
    const int scol = (tid & 15) << 2;   // 0,4,...,60
    const int soff = srow * KCH + scol;
    const float* xsrc = X + (size_t)(m0 + srow) * LEAF + scol;

    const int NFULL = LEAF / KCH;       // 76 full chunks (4864)
    v8f acc = {};

    // prologue: stage chunk 0
    {
        float4 v = *(const float4*)(xsrc);
        *(v4h*)&aT[0][soff] =
            (v4h){(_Float16)v.x, (_Float16)v.y, (_Float16)v.z, (_Float16)v.w};
    }
    __syncthreads();

    for (int c = 0; c < NFULL; ++c) {
        const int cur = c & 1;
        const bool have_next = (c + 1 < NFULL);
        float4 nv;
        if (have_next) nv = *(const float4*)(xsrc + (size_t)(c + 1) * KCH);

        const int kg = c * KCH;
        v16h a0 = load_a_frag_v(&aT[cur][0], KCH, 0, 0);
        v16h b0 = load_b_frag_v(Wkn, ATTN, kg, n0);
        acc = WMMA_F16(a0, b0, acc);
        v16h a1 = load_a_frag_v(&aT[cur][0], KCH, 0, 32);
        v16h b1 = load_b_frag_v(Wkn, ATTN, kg + 32, n0);
        acc = WMMA_F16(a1, b1, acc);

        if (have_next)
            *(v4h*)&aT[cur ^ 1][soff] =
                (v4h){(_Float16)nv.x, (_Float16)nv.y, (_Float16)nv.z, (_Float16)nv.w};
        __syncthreads();
    }

    // K tail: 4864..4879 (16 valid cols, pad to 32)
    {
        const int kg = NFULL * KCH;
        for (int i = tid; i < 16 * 32; i += 256) {
            int r = i >> 5, cc = i & 31;
            int k = kg + cc;
            aT[0][r * KCH + cc] =
                (k < LEAF) ? (_Float16)X[(size_t)(m0 + r) * LEAF + k]
                           : (_Float16)0.f;
        }
        __syncthreads();
        v16h a = load_a_frag_v(&aT[0][0], KCH, 0, 0);
        v16h b = load_b_frag_chk(Wkn, ATTN, kg, n0, LEAF);
        acc = WMMA_F16(a, b, acc);
    }

    const int l  = tid & 31;
    const int n  = n0 + (l & 15);
    const int mb = m0 + ((l >> 4) << 3);
#pragma unroll
    for (int v = 0; v < 8; ++v)
        out[(size_t)(mb + v) * ATTN + n] = (_Float16)tanhf(acc[v]);
}

// ---------------------------------------------------------------------------
// 2) C[M,N] = A_f16[M,K] @ BT_f16[K,N] + bias  -> f32   (GRU input gates, etc.)
// ---------------------------------------------------------------------------
__global__ void gemm_kn_bias_kernel(const _Float16* __restrict__ A,
                                    const _Float16* __restrict__ BT,
                                    const float* __restrict__ bias,
                                    float* __restrict__ C,
                                    int M, int N, int K) {
    const int nt_n = N >> 4;
    const int tile = blockIdx.x * (blockDim.x >> 5) + (threadIdx.x >> 5);
    const int m0 = (tile / nt_n) << 4;
    const int n0 = (tile % nt_n) << 4;
    v8f acc = {};
    for (int k0 = 0; k0 < K; k0 += 32) {
        v16h a = load_a_frag_v(A, K, m0, k0);
        v16h b = load_b_frag_v(BT, N, k0, n0);
        acc = WMMA_F16(a, b, acc);
    }
    const int l  = threadIdx.x & 31;
    const int n  = n0 + (l & 15);
    const int mb = m0 + ((l >> 4) << 3);
    const float bv = bias[n];
#pragma unroll
    for (int v = 0; v < 8; ++v)
        C[(size_t)(mb + v) * N + n] = acc[v] + bv;
}

// ---------------------------------------------------------------------------
// 3) GRU layer: 8 blocks x 16 batch rows; recurrent matmul via WMMA from LDS
// ---------------------------------------------------------------------------
__global__ void gru_layer_kernel(const float* __restrict__ xW,      // [M, 384]
                                 const _Float16* __restrict__ WhhT, // [128,384] f16
                                 const float* __restrict__ bhh,     // [384]
                                 float* __restrict__ out_f,         // [M,128]
                                 _Float16* __restrict__ out_h) {    // [M,128]
    __shared__ _Float16 h_h[16 * RNN];
    __shared__ float    h_f[16 * RNN];
    __shared__ float    gh[16 * G3];
    const int b0   = blockIdx.x * 16;
    const int wave = threadIdx.x >> 5;
    for (int i = threadIdx.x; i < 16 * RNN; i += 256) {
        h_h[i] = (_Float16)0.f;
        h_f[i] = 0.f;
    }
    __syncthreads();
    for (int t = 0; t < TT; ++t) {
        // gh = h @ Whh^T + bhh : 24 N-tiles over 8 waves
        for (int nt = wave; nt < (G3 >> 4); nt += 8) {
            const int n0 = nt << 4;
            v8f acc = {};
#pragma unroll
            for (int k0 = 0; k0 < RNN; k0 += 32) {
                v16h a = load_a_frag_v(h_h, RNN, 0, k0);
                v16h b = load_b_frag_v(WhhT, G3, k0, n0);
                acc = WMMA_F16(a, b, acc);
            }
            const int l  = threadIdx.x & 31;
            const int n  = n0 + (l & 15);
            const int mb = (l >> 4) << 3;
            const float bv = bhh[n];
#pragma unroll
            for (int v = 0; v < 8; ++v) gh[(mb + v) * G3 + n] = acc[v] + bv;
        }
        __syncthreads();
        // gates (2048 elements / 256 threads)
        for (int i = threadIdx.x; i < 16 * RNN; i += 256) {
            const int r = i >> 7, j = i & 127;
            const size_t mrow = (size_t)(b0 + r) * TT + t;
            const float* xwp = xW + mrow * G3;
            float xr = xwp[j], xz = xwp[RNN + j], xn = xwp[2 * RNN + j];
            float hr = gh[r * G3 + j];
            float hz = gh[r * G3 + RNN + j];
            float hn = gh[r * G3 + 2 * RNN + j];
            float rg = sigmoidf_(xr + hr);
            float zg = sigmoidf_(xz + hz);
            float ng = tanhf(xn + rg * hn);
            float hv = (1.f - zg) * ng + zg * h_f[i];
            h_f[i] = hv;
            h_h[i] = (_Float16)hv;
            out_f[mrow * RNN + j] = hv;
            out_h[mrow * RNN + j] = (_Float16)hv;
        }
        __syncthreads();
    }
}

// ---------------------------------------------------------------------------
// 4) ancestor attention: per (b,t), 64 threads; builds s = [out, k] in f16
// ---------------------------------------------------------------------------
__global__ void attn_kernel(const int* __restrict__ inputs_f,
                            const float* __restrict__ embed_a,  // [729,128]
                            const float* __restrict__ out_f,
                            const _Float16* __restrict__ out_h,
                            _Float16* __restrict__ s_h) {       // [M,256]
    __shared__ float ov[ATTN];
    __shared__ float lsh[AA][RNN];
    __shared__ float wgt[AA];
    __shared__ float red[AA];
    const int m   = blockIdx.x;
    const int tid = threadIdx.x;  // 0..63
    ov[tid]      = out_f[(size_t)m * RNN + tid];
    ov[tid + 64] = out_f[(size_t)m * RNN + 64 + tid];
    __syncthreads();
    const int idx = inputs_f[(size_t)m * AA + tid];
    const float* lrow = embed_a + (size_t)idx * RNN;
    float dot = 0.f;
#pragma unroll 8
    for (int j = 0; j < RNN; ++j) {
        float lv = lrow[j];
        lsh[tid][j] = lv;
        dot += ov[j] * lv;
    }
    float hl = dot + ((idx > 0) ? 0.f : -1e30f);
    red[tid] = hl;
    __syncthreads();
    for (int s = 32; s > 0; s >>= 1) {
        if (tid < s) red[tid] = fmaxf(red[tid], red[tid + s]);
        __syncthreads();
    }
    float mx = red[0];
    __syncthreads();
    float e = __expf(hl - mx);
    red[tid] = e;
    __syncthreads();
    for (int s = 32; s > 0; s >>= 1) {
        if (tid < s) red[tid] += red[tid + s];
        __syncthreads();
    }
    wgt[tid] = e / red[0];
    __syncthreads();
#pragma unroll
    for (int jj = 0; jj < 2; ++jj) {
        const int j = tid + jj * 64;
        float acc = 0.f;
#pragma unroll 8
        for (int a = 0; a < AA; ++a) acc += wgt[a] * lsh[a][j];
        s_h[(size_t)m * (2 * RNN) + RNN + j] = (_Float16)acc;
        s_h[(size_t)m * (2 * RNN) + j]       = out_h[(size_t)m * RNN + j];
    }
}

// ---------------------------------------------------------------------------
// 5) fc: sigmoid(s @ fc_w^T + fc_b) * mask  -> d_out [M,283]
//    B pre-transposed+padded to [256, 288] (zero cols), no in-loop checks.
// ---------------------------------------------------------------------------
__global__ void fc_kernel(const _Float16* __restrict__ S,      // [M,256]
                          const _Float16* __restrict__ WfcT,   // [256,288]
                          const float* __restrict__ bias,      // [283]
                          const float* __restrict__ mask,      // [M]
                          float* __restrict__ out) {           // [M,283]
    const int K = 2 * RNN, nt_n = NCP >> 4;
    const int tile = blockIdx.x * (blockDim.x >> 5) + (threadIdx.x >> 5);
    const int m0 = (tile / nt_n) << 4;
    const int n0 = (tile % nt_n) << 4;
    v8f acc = {};
#pragma unroll
    for (int k0 = 0; k0 < K; k0 += 32) {
        v16h a = load_a_frag_v(S, K, m0, k0);
        v16h b = load_b_frag_v(WfcT, NCP, k0, n0);
        acc = WMMA_F16(a, b, acc);
    }
    const int l  = threadIdx.x & 31;
    const int n  = n0 + (l & 15);
    const int mb = m0 + ((l >> 4) << 3);
    if (n < NC) {
        const float bv = bias[n];
#pragma unroll
        for (int v = 0; v < 8; ++v) {
            const int m = mb + v;
            out[(size_t)m * NC + n] = sigmoidf_(acc[v] + bv) * mask[m];
        }
    }
}

// ---------------------------------------------------------------------------
extern "C" void kernel_launch(void* const* d_in, const int* in_sizes, int n_in,
                              void* d_out, int out_size, void* d_ws, size_t ws_size,
                              hipStream_t stream) {
    const float* inputs_x = (const float*)d_in[0];
    const int*   inputs_f = (const int*)d_in[1];
    const float* mask     = (const float*)d_in[2];
    const float* dag_emb  = (const float*)d_in[3];
    const float* embed_a  = (const float*)d_in[4];
    const float* Wih0     = (const float*)d_in[5];
    const float* Whh0     = (const float*)d_in[6];
    const float* bih0     = (const float*)d_in[7];
    const float* bhh0     = (const float*)d_in[8];
    const float* Wih1     = (const float*)d_in[9];
    const float* Whh1     = (const float*)d_in[10];
    const float* bih1     = (const float*)d_in[11];
    const float* bhh1     = (const float*)d_in[12];
    const float* fc_w     = (const float*)d_in[13];
    const float* fc_b     = (const float*)d_in[14];
    float* out = (float*)d_out;

    // workspace carve-out (256B aligned)
    char* wp = (char*)d_ws;
    auto alloc = [&](size_t bytes) {
        char* p = wp;
        wp += (bytes + 255) & ~(size_t)255;
        return p;
    };
    _Float16* dag_h   = (_Float16*)alloc((size_t)LEAF * ATTN * 2);   // [4880,128]
    _Float16* Wih0T   = (_Float16*)alloc((size_t)ATTN * G3 * 2);     // [128,384]
    _Float16* Whh0T   = (_Float16*)alloc((size_t)RNN * G3 * 2);      // [128,384]
    _Float16* Wih1T   = (_Float16*)alloc((size_t)RNN * G3 * 2);
    _Float16* Whh1T   = (_Float16*)alloc((size_t)RNN * G3 * 2);
    _Float16* fcwT    = (_Float16*)alloc((size_t)(2 * RNN) * NCP * 2); // [256,288]
    _Float16* x_h     = (_Float16*)alloc((size_t)MM * ATTN * 2);
    float*    xw      = (float*)   alloc((size_t)MM * G3 * 4);       // reused per layer
    float*    h1_f    = (float*)   alloc((size_t)MM * RNN * 4);
    _Float16* h1_h    = (_Float16*)alloc((size_t)MM * RNN * 2);
    float*    o2_f    = (float*)   alloc((size_t)MM * RNN * 4);
    _Float16* o2_h    = (_Float16*)alloc((size_t)MM * RNN * 2);
    _Float16* s_h     = (_Float16*)alloc((size_t)MM * 2 * RNN * 2);

    // weight conversion (+ transpose so every WMMA B operand is [K,N] contiguous)
    cvt_f32_f16_kernel<<<(LEAF * ATTN + 255) / 256, 256, 0, stream>>>(
        dag_emb, dag_h, LEAF * ATTN);
    auto cvtT = [&](const float* s, _Float16* d, int N, int K, int Npad) {
        cvt_transpose_kernel<<<(K * Npad + 255) / 256, 256, 0, stream>>>(
            s, d, N, K, Npad);
    };
    cvtT(Wih0, Wih0T, G3, ATTN, G3);
    cvtT(Whh0, Whh0T, G3, RNN, G3);
    cvtT(Wih1, Wih1T, G3, RNN, G3);
    cvtT(Whh1, Whh1T, G3, RNN, G3);
    cvtT(fc_w, fcwT, NC, 2 * RNN, NCP);

    // 1) projection + tanh
    proj_tanh_kernel<<<MM / 16, 256, 0, stream>>>(inputs_x, dag_h, x_h);

    // 2) layer-0 input gates, 3) layer-0 GRU
    gemm_kn_bias_kernel<<<(MM / 16) * (G3 / 16) / 8, 256, 0, stream>>>(
        x_h, Wih0T, bih0, xw, MM, G3, ATTN);
    gru_layer_kernel<<<BB / 16, 256, 0, stream>>>(xw, Whh0T, bhh0, h1_f, h1_h);

    // 4) layer-1 input gates, 5) layer-1 GRU
    gemm_kn_bias_kernel<<<(MM / 16) * (G3 / 16) / 8, 256, 0, stream>>>(
        h1_h, Wih1T, bih1, xw, MM, G3, RNN);
    gru_layer_kernel<<<BB / 16, 256, 0, stream>>>(xw, Whh1T, bhh1, o2_f, o2_h);

    // 6) attention -> s = [out, k]
    attn_kernel<<<MM, 64, 0, stream>>>(inputs_f, embed_a, o2_f, o2_h, s_h);

    // 7) fc + sigmoid + mask
    fc_kernel<<<(MM / 16) * (NCP / 16) / 8, 256, 0, stream>>>(
        s_h, fcwT, fc_b, mask, out);
}
